// GATModel_35966056136909
// MI455X (gfx1250) — compile-verified, matrix operand order checked by reading, not run on hardware
//
#include <hip/hip_runtime.h>
#include <hip/hip_bf16.h>

// ---------------- problem constants (from reference) ----------------
#define NN     16384
#define EE     262144
#define EP     (EE + NN)        // edges + self loops = 278528
#define IN_F   3
#define HID    128
#define ENC    768
#define NEG_SLOPE 0.2f
#define LN_EPS    1e-5f

typedef __attribute__((ext_vector_type(16))) __bf16 bf16x16;
typedef __attribute__((ext_vector_type(8)))  float  f32x8;

// ---------------- small device helpers ----------------
__device__ __forceinline__ float waveReduceSum(float v) {
#pragma unroll
  for (int off = 16; off > 0; off >>= 1) v += __shfl_xor(v, off, 32);
  return v;
}

// monotone encoding of float -> uint so atomicMax(u32) implements float max
__device__ __forceinline__ unsigned encf(float f) {
  unsigned u = __float_as_uint(f);
  return (u >> 31) ? ~u : (u | 0x80000000u);
}
__device__ __forceinline__ float decf(unsigned e) {
  return (e & 0x80000000u) ? __uint_as_float(e ^ 0x80000000u)
                           : __uint_as_float(~e);
}

__device__ __forceinline__ unsigned short f2bf(float f) {
  unsigned u = __float_as_uint(f);
  unsigned r = u + 0x7fffu + ((u >> 16) & 1u);   // round to nearest even
  return (unsigned short)(r >> 16);
}

__device__ __forceinline__ float lrelu(float v) {
  return v > 0.f ? v : NEG_SLOPE * v;
}

// ---------------- fills ----------------
__global__ void fill_f32(float* p, float v, int n) {
  int i = blockIdx.x * blockDim.x + threadIdx.x;
  if (i < n) p[i] = v;
}
__global__ void fill_u32(unsigned* p, unsigned v, int n) {
  int i = blockIdx.x * blockDim.x + threadIdx.x;
  if (i < n) p[i] = v;
}

// ---------------- res = x @ Wfc + bfc   [N,3]x[3,768] ----------------
__global__ void fc_kernel(const float* __restrict__ x, const float* __restrict__ W,
                          const float* __restrict__ b, float* __restrict__ res) {
  int i = blockIdx.x * blockDim.x + threadIdx.x;
  if (i >= NN * ENC) return;
  int n = i / ENC, c = i % ENC;
  float acc = b[c];
#pragma unroll
  for (int k = 0; k < IN_F; ++k) acc += x[n * IN_F + k] * W[k * ENC + c];
  res[i] = acc;
}

// ---------------- layer1 linears: xl1/xr1 = x@W + b  [N,3]x[3,128] ----------------
__global__ void lin1_kernel(const float* __restrict__ x,
                            const float* __restrict__ Wl, const float* __restrict__ bl,
                            const float* __restrict__ Wr, const float* __restrict__ br,
                            float* __restrict__ xl, float* __restrict__ xr) {
  int i = blockIdx.x * blockDim.x + threadIdx.x;
  if (i >= NN * HID) return;
  int n = i / HID, c = i % HID;
  float al = bl[c], ar = br[c];
#pragma unroll
  for (int k = 0; k < IN_F; ++k) {
    float xv = x[n * IN_F + k];
    al += xv * Wl[k * HID + c];
    ar += xv * Wr[k * HID + c];
  }
  xl[i] = al;
  xr[i] = ar;
}

// ---------------- edge scores, one wave per edge, float4 (b128) gathers ----------------
// lane handles channels c = j*128 + lane*4 .. +3 for j in [0, C/128)
template <int C>
__global__ void edge_score_kernel(const float* __restrict__ xl, const float* __restrict__ xr,
                                  const float* __restrict__ att,
                                  const int* __restrict__ src, const int* __restrict__ dst,
                                  float* __restrict__ esc) {
  int wid  = (blockIdx.x * blockDim.x + threadIdx.x) >> 5;
  int lane = threadIdx.x & 31;
  if (wid >= EP) return;
  int s, d;
  if (wid < EE) { s = src[wid]; d = dst[wid]; } else { s = d = wid - EE; }
  float acc = 0.f;
#pragma unroll
  for (int j = 0; j < C / 128; ++j) {
    int c0 = j * 128 + lane * 4;
    float4 a = *(const float4*)(xl + (size_t)s * C + c0);
    float4 b = *(const float4*)(xr + (size_t)d * C + c0);
    float4 w = *(const float4*)(att + c0);
    acc += lrelu(a.x + b.x) * w.x + lrelu(a.y + b.y) * w.y +
           lrelu(a.z + b.z) * w.z + lrelu(a.w + b.w) * w.w;
  }
  acc = waveReduceSum(acc);
  if (lane == 0) esc[wid] = acc;
}

// ---------------- per-dst max (atomicMax on monotone uint) ----------------
__global__ void edge_max_kernel(const float* __restrict__ esc,
                                const int* __restrict__ dst, unsigned* __restrict__ menc) {
  int e = blockIdx.x * blockDim.x + threadIdx.x;
  if (e >= EP) return;
  int d = (e < EE) ? dst[e] : e - EE;
  atomicMax(&menc[d], encf(esc[e]));
}

// ---------------- exp + per-dst denom ----------------
__global__ void edge_exp_kernel(float* __restrict__ esc, const int* __restrict__ dst,
                                const unsigned* __restrict__ menc, float* __restrict__ denom) {
  int e = blockIdx.x * blockDim.x + threadIdx.x;
  if (e >= EP) return;
  int d = (e < EE) ? dst[e] : e - EE;
  float ex = expf(esc[e] - decf(menc[d]));
  esc[e] = ex;
  atomicAdd(&denom[d], ex);
}

// ---- weighted scatter: agg[dst] += alpha * xl[src], wave/edge, b128 gathers ----
template <int C>
__global__ void edge_agg_kernel(const float* __restrict__ esc, const float* __restrict__ denom,
                                const float* __restrict__ xl,
                                const int* __restrict__ src, const int* __restrict__ dst,
                                float* __restrict__ agg) {
  int wid  = (blockIdx.x * blockDim.x + threadIdx.x) >> 5;
  int lane = threadIdx.x & 31;
  if (wid >= EP) return;
  int s, d;
  if (wid < EE) { s = src[wid]; d = dst[wid]; } else { s = d = wid - EE; }
  float alpha = esc[wid] / denom[d];
#pragma unroll
  for (int j = 0; j < C / 128; ++j) {
    int c0 = j * 128 + lane * 4;
    float4 a = *(const float4*)(xl + (size_t)s * C + c0);
    float* p = agg + (size_t)d * C + c0;
    atomicAdd(p + 0, alpha * a.x);
    atomicAdd(p + 1, alpha * a.y);
    atomicAdd(p + 2, alpha * a.z);
    atomicAdd(p + 3, alpha * a.w);
  }
}

// ---------------- bias + layernorm + elu -> bf16 h, wave per node (HID=128) ----------------
__global__ void ln1_elu_kernel(const float* __restrict__ agg, const float* __restrict__ bias,
                               const float* __restrict__ g, const float* __restrict__ b,
                               unsigned short* __restrict__ hbf) {
  int n    = (blockIdx.x * blockDim.x + threadIdx.x) >> 5;
  int lane = threadIdx.x & 31;
  if (n >= NN) return;
  int c0 = lane * 4;
  float4 v  = *(const float4*)(agg + (size_t)n * HID + c0);
  float4 bb = *(const float4*)(bias + c0);
  v.x += bb.x; v.y += bb.y; v.z += bb.z; v.w += bb.w;
  float mu = waveReduceSum(v.x + v.y + v.z + v.w) * (1.0f / HID);
  float dx = v.x - mu, dy = v.y - mu, dz = v.z - mu, dw = v.w - mu;
  float var = waveReduceSum(dx * dx + dy * dy + dz * dz + dw * dw) * (1.0f / HID);
  float rs = rsqrtf(var + LN_EPS);
  float4 gg = *(const float4*)(g + c0);
  float4 be = *(const float4*)(b + c0);
  float y0 = dx * rs * gg.x + be.x;
  float y1 = dy * rs * gg.y + be.y;
  float y2 = dz * rs * gg.z + be.z;
  float y3 = dw * rs * gg.w + be.w;
  y0 = y0 > 0.f ? y0 : expf(y0) - 1.f;
  y1 = y1 > 0.f ? y1 : expf(y1) - 1.f;
  y2 = y2 > 0.f ? y2 : expf(y2) - 1.f;
  y3 = y3 > 0.f ? y3 : expf(y3) - 1.f;
  ushort4 h4;
  h4.x = f2bf(y0); h4.y = f2bf(y1); h4.z = f2bf(y2); h4.w = f2bf(y3);
  *(ushort4*)(hbf + (size_t)n * HID + c0) = h4;
}

// ---------------- W[128,768] fp32 row-major -> bf16 column-major Wt[768,128] ----------------
__global__ void convert_wt_kernel(const float* __restrict__ W, unsigned short* __restrict__ Wt) {
  int i = blockIdx.x * blockDim.x + threadIdx.x;
  if (i >= HID * ENC) return;
  int k = i / ENC, n = i % ENC;
  Wt[n * HID + k] = f2bf(W[i]);
}

// ---------------- WMMA GEMM: C[N,768] = A[N,128](bf16) x W(bf16) + bias -----------------
// One wave computes a 64x16 output tile: 4 f32x8 accumulators, B fragment reused across
// 4 v_wmma_f32_16x16x32_bf16 issues per k-step; K=128 -> 16 WMMA per wave.
// A-fragment (ISA 7.12.2, 16-bit A 16x32): lane L holds row M=L&15; element i maps to
// K = kb + (L>=16 ? 8 : 0) + (i<8 ? i : 16 + i-8)  -> two contiguous 16B loads per frag.
// B loads mirror A because Wt is stored column-major.
__global__ void __launch_bounds__(256)
wmma_gemm_kernel(const unsigned short* __restrict__ A,   // [N,128] bf16 row-major
                 const unsigned short* __restrict__ Bt,  // [768,128] bf16 (col-major W)
                 const float* __restrict__ bias,         // [768]
                 float* __restrict__ C) {                // [N,768]
  const int nTilesN = ENC / 16;                          // 48
  int wave = threadIdx.x >> 5;
  int lane = threadIdx.x & 31;
  int tile = blockIdx.x * 8 + wave;                      // 64x16 macro-tiles
  int tm = tile / nTilesN;                               // 0..255
  int tn = tile - tm * nTilesN;
  if (tm * 64 >= NN) return;                             // uniform per wave

  int half = lane >> 4;                                  // 0|1
  int l15  = lane & 15;
  int koff = half * 8;

  const unsigned short* a0   = A  + (size_t)(tm * 64 + l15) * HID;
  const unsigned short* bcol = Bt + (size_t)(tn * 16 + l15) * HID;

  f32x8 acc[4];
#pragma unroll
  for (int t = 0; t < 4; ++t) acc[t] = (f32x8){0.f,0.f,0.f,0.f,0.f,0.f,0.f,0.f};

#pragma unroll
  for (int kb = 0; kb < HID; kb += 32) {
    union { uint4 u[2]; bf16x16 v; } fb;
    fb.u[0] = *(const uint4*)(bcol + kb + koff);
    fb.u[1] = *(const uint4*)(bcol + kb + 16 + koff);
#pragma unroll
    for (int t = 0; t < 4; ++t) {
      const unsigned short* ar = a0 + (size_t)t * 16 * HID;
      union { uint4 u[2]; bf16x16 v; } fa;
      fa.u[0] = *(const uint4*)(ar + kb + koff);
      fa.u[1] = *(const uint4*)(ar + kb + 16 + koff);
      acc[t] = __builtin_amdgcn_wmma_f32_16x16x32_bf16(false, fa.v, false, fb.v,
                                                       (short)0, acc[t], false, false);
    }
  }
  int col = tn * 16 + l15;
  float bv = bias[col];
#pragma unroll
  for (int t = 0; t < 4; ++t) {
#pragma unroll
    for (int r = 0; r < 8; ++r) {
      int m = tm * 64 + t * 16 + r + half * 8;           // C/D layout: vgpr r, half -> M+8
      C[(size_t)m * ENC + col] = acc[t][r] + bv;
    }
  }
}

// ------- final: out = LN(out + bias2 + res) * g + b, wave per node, b128 loads -------
__global__ void final_ln_kernel(const float* __restrict__ res, const float* __restrict__ bias2,
                                const float* __restrict__ g, const float* __restrict__ b,
                                float* __restrict__ out) {
  int n    = (blockIdx.x * blockDim.x + threadIdx.x) >> 5;
  int lane = threadIdx.x & 31;
  if (n >= NN) return;
  float4 v[ENC / 128];
  float s = 0.f;
#pragma unroll
  for (int j = 0; j < ENC / 128; ++j) {
    int c0 = j * 128 + lane * 4;
    float4 o  = *(const float4*)(out + (size_t)n * ENC + c0);
    float4 bb = *(const float4*)(bias2 + c0);
    float4 rr = *(const float4*)(res + (size_t)n * ENC + c0);
    v[j].x = o.x + bb.x + rr.x;
    v[j].y = o.y + bb.y + rr.y;
    v[j].z = o.z + bb.z + rr.z;
    v[j].w = o.w + bb.w + rr.w;
    s += v[j].x + v[j].y + v[j].z + v[j].w;
  }
  float mu = waveReduceSum(s) * (1.0f / ENC);
  float q = 0.f;
#pragma unroll
  for (int j = 0; j < ENC / 128; ++j) {
    float dx = v[j].x - mu, dy = v[j].y - mu, dz = v[j].z - mu, dw = v[j].w - mu;
    q += dx * dx + dy * dy + dz * dz + dw * dw;
  }
  float var = waveReduceSum(q) * (1.0f / ENC);
  float rs = rsqrtf(var + LN_EPS);
#pragma unroll
  for (int j = 0; j < ENC / 128; ++j) {
    int c0 = j * 128 + lane * 4;
    float4 gg = *(const float4*)(g + c0);
    float4 be = *(const float4*)(b + c0);
    float4 y;
    y.x = (v[j].x - mu) * rs * gg.x + be.x;
    y.y = (v[j].y - mu) * rs * gg.y + be.y;
    y.z = (v[j].z - mu) * rs * gg.z + be.z;
    y.w = (v[j].w - mu) * rs * gg.w + be.w;
    *(float4*)(out + (size_t)n * ENC + c0) = y;
  }
}

// ====================================================================
extern "C" void kernel_launch(void* const* d_in, const int* in_sizes, int n_in,
                              void* d_out, int out_size, void* d_ws, size_t ws_size,
                              hipStream_t stream) {
  (void)in_sizes; (void)n_in; (void)out_size; (void)ws_size;

  const float* x      = (const float*)d_in[0];
  const int*   eidx   = (const int*)d_in[1];      // [2,E] (JAX default int32)
  const float* Wfc    = (const float*)d_in[2];
  const float* bfc    = (const float*)d_in[3];
  const float* W1l    = (const float*)d_in[4];
  const float* b1l    = (const float*)d_in[5];
  const float* W1r    = (const float*)d_in[6];
  const float* b1r    = (const float*)d_in[7];
  const float* att1   = (const float*)d_in[8];
  const float* bias1  = (const float*)d_in[9];
  const float* g_bn1  = (const float*)d_in[10];
  const float* bb_bn1 = (const float*)d_in[11];
  const float* W2l    = (const float*)d_in[12];
  const float* b2l    = (const float*)d_in[13];
  const float* W2r    = (const float*)d_in[14];
  const float* b2r    = (const float*)d_in[15];
  const float* att2   = (const float*)d_in[16];
  const float* bias2  = (const float*)d_in[17];
  const float* g_norm = (const float*)d_in[18];
  const float* b_norm = (const float*)d_in[19];

  const int* src = eidx;
  const int* dst = eidx + EE;
  float* out = (float*)d_out;

  // ---- workspace carving (256B aligned) ----
  char* ws = (char*)d_ws;
  size_t off = 0;
  auto carve = [&](size_t bytes) -> char* {
    char* p = ws + off;
    off += (bytes + 255) & ~(size_t)255;
    return p;
  };
  float*          resbuf = (float*)         carve((size_t)NN * ENC * 4);
  float*          xl1    = (float*)         carve((size_t)NN * HID * 4);
  float*          xr1    = (float*)         carve((size_t)NN * HID * 4);
  float*          agg1   = (float*)         carve((size_t)NN * HID * 4);
  unsigned short* hbf    = (unsigned short*)carve((size_t)NN * HID * 2);
  unsigned short* WtL    = (unsigned short*)carve((size_t)HID * ENC * 2);
  unsigned short* WtR    = (unsigned short*)carve((size_t)HID * ENC * 2);
  float*          xl2    = (float*)         carve((size_t)NN * ENC * 4);
  float*          xr2    = (float*)         carve((size_t)NN * ENC * 4);
  float*          esc    = (float*)         carve((size_t)EP * 4);
  unsigned*       menc   = (unsigned*)      carve((size_t)NN * 4);
  float*          denom  = (float*)         carve((size_t)NN * 4);

  const int T = 256;
  auto blk = [](long n, int t) { return (int)((n + t - 1) / t); };
  const int edgeWaveBlocks = blk((long)EP * 32, T);
  const int nodeWaveBlocks = blk((long)NN * 32, T);

  // ---- residual projection ----
  fc_kernel<<<blk((long)NN * ENC, T), T, 0, stream>>>(x, Wfc, bfc, resbuf);

  // ---- GATv2 layer 1 (HID=128) ----
  lin1_kernel<<<blk((long)NN * HID, T), T, 0, stream>>>(x, W1l, b1l, W1r, b1r, xl1, xr1);
  fill_u32<<<blk(NN, T), T, 0, stream>>>(menc, 0u, NN);
  fill_f32<<<blk(NN, T), T, 0, stream>>>(denom, 0.f, NN);
  fill_f32<<<blk((long)NN * HID, T), T, 0, stream>>>(agg1, 0.f, NN * HID);
  edge_score_kernel<HID><<<edgeWaveBlocks, T, 0, stream>>>(xl1, xr1, att1, src, dst, esc);
  edge_max_kernel<<<blk(EP, T), T, 0, stream>>>(esc, dst, menc);
  edge_exp_kernel<<<blk(EP, T), T, 0, stream>>>(esc, dst, menc, denom);
  edge_agg_kernel<HID><<<edgeWaveBlocks, T, 0, stream>>>(esc, denom, xl1, src, dst, agg1);
  ln1_elu_kernel<<<nodeWaveBlocks, T, 0, stream>>>(agg1, bias1, g_bn1, bb_bn1, hbf);

  // ---- GATv2 layer 2 (ENC=768): WMMA bf16 GEMMs for xl2/xr2 ----
  convert_wt_kernel<<<blk((long)HID * ENC, T), T, 0, stream>>>(W2l, WtL);
  convert_wt_kernel<<<blk((long)HID * ENC, T), T, 0, stream>>>(W2r, WtR);
  const int gemmBlocks = (NN / 64) * (ENC / 16) / 8;     // 8 waves (64x16 tiles) per block
  wmma_gemm_kernel<<<gemmBlocks, T, 0, stream>>>(hbf, WtL, b2l, xl2);
  wmma_gemm_kernel<<<gemmBlocks, T, 0, stream>>>(hbf, WtR, b2r, xr2);

  fill_u32<<<blk(NN, T), T, 0, stream>>>(menc, 0u, NN);
  fill_f32<<<blk(NN, T), T, 0, stream>>>(denom, 0.f, NN);
  fill_f32<<<blk((long)NN * ENC, T), T, 0, stream>>>(out, 0.f, NN * ENC);
  edge_score_kernel<ENC><<<edgeWaveBlocks, T, 0, stream>>>(xl2, xr2, att2, src, dst, esc);
  edge_max_kernel<<<blk(EP, T), T, 0, stream>>>(esc, dst, menc);
  edge_exp_kernel<<<blk(EP, T), T, 0, stream>>>(esc, dst, menc, denom);
  edge_agg_kernel<ENC><<<edgeWaveBlocks, T, 0, stream>>>(esc, denom, xl2, src, dst, out);

  // ---- bias2 + residual + final layernorm, fused in-place on d_out ----
  final_ln_kernel<<<nodeWaveBlocks, T, 0, stream>>>(resbuf, bias2, g_norm, b_norm, out);
}
